// TransformerBlock_39427799777555
// MI455X (gfx1250) — matrix operand order repaired and checked by reading, not compiled, not run on hardware
//
#include <hip/hip_runtime.h>
#include <hip/hip_bf16.h>

// ---------------------------------------------------------------------------
// Types for CDNA5 WMMA (wave32): D(16x16 f32) = A(16x32 bf16) * B(32x16 bf16) + C
// ---------------------------------------------------------------------------
typedef __attribute__((ext_vector_type(16))) __bf16 v16bf;
typedef __attribute__((ext_vector_type(8)))  float  v8f;

#define BM 64
#define BN 128
#define BK 32
#define BKP (BK + 8)   // padded K pitch (40 halves = 80B, 16B-aligned rows)

union FragU {
    v16bf v;
    uint4 q[2];
};

// ---------------------------------------------------------------------------
// Generic batched bf16 GEMM with fused epilogue.
//   C[M,N] = A[M,Kd] * B[Kd,N]  (+ bias[n]) (+ residual[m,n]) (relu)
// A,B row-major bf16; Kd must be a multiple of 32.
// Per workgroup: 64x128 tile, 8 waves, each wave 32x32 = four 16x16 WMMA tiles.
// A staged row-major [M][K]; B staged TRANSPOSED [N][K] so both A and B
// fragments are two contiguous ds_load_b128 per 16x16 tile.
// Fragment layouts follow CDNA5 ISA §7.12.2 (wave32).
// ---------------------------------------------------------------------------
__global__ __launch_bounds__(256)
void gemm_bf16_kernel(const __hip_bfloat16* __restrict__ A, long long sA, int lda,
                      const __hip_bfloat16* __restrict__ B, long long sB, int ldb,
                      float* __restrict__ Cf, long long sCf, int ldcf,
                      __hip_bfloat16* __restrict__ Cb, long long sCb, int ldcb,
                      const float* __restrict__ bias,
                      const float* __restrict__ res, long long sR, int ldr,
                      int M, int N, int Kd, int flags)
{
    __shared__ __bf16 As [BM][BKP];   // row-major A tile
    __shared__ __bf16 Bst[BN][BKP];   // TRANSPOSED B tile: [n][k]

    const int tid   = threadIdx.x;
    const int lane  = tid & 31;
    const int wid   = tid >> 5;
    const int waveM = wid >> 2;           // 0..1
    const int waveN = wid & 3;            // 0..3
    const int batch = blockIdx.z;
    const int m0    = blockIdx.y * BM;
    const int n0    = blockIdx.x * BN;

    const __bf16* Ab = (const __bf16*)A + (size_t)batch * (size_t)sA;
    const __bf16* Bb = (const __bf16*)B + (size_t)batch * (size_t)sB;

    v8f acc[2][2] = {};

    // A staging: 4 threads per row (8 halves / 16B each), 64 rows
    const int arow = tid >> 2;
    const int acol = (tid & 3) * 8;
    // B staging: each thread covers 2 columns x 8 rows of the 32x128 tile
    const int tcol = (tid & 63) * 2;      // 0..126
    const int trow = (tid >> 6) * 8;      // 0,8,16,24

    const int halfsel = lane >> 4;   // 0 or 1
    const int sublane = lane & 15;

    // fast path: dword-aligned global rows and full-width N tile
    const bool bfast = ((ldb & 1) == 0) && (n0 + BN <= N);

    for (int k0 = 0; k0 < Kd; k0 += BK) {
        // ---- stage A tile (row-major) ----
        {
            const int gm = m0 + arow;
            if (gm < M) {
                const __bf16* src = Ab + (size_t)gm * lda + (k0 + acol);
                *(uint4*)&As[arow][acol] = *(const uint4*)src;
                if (k0 + BK < Kd) __builtin_prefetch(src + BK, 0, 3);
            } else {
                uint4 z = {0u, 0u, 0u, 0u};
                *(uint4*)&As[arow][acol] = z;
            }
        }
        // ---- stage B tile transposed: Bst[n][k] = B[k0+k][n0+n] ----
        {
            const __bf16* src = Bb + (size_t)(k0 + trow) * ldb + (n0 + tcol);
            if (bfast) {
                uint r[8];
                #pragma unroll
                for (int e = 0; e < 8; ++e)
                    r[e] = *(const uint*)(src + (size_t)e * ldb);
                if (k0 + BK < Kd) __builtin_prefetch(src + (size_t)BK * ldb, 0, 3);
                uint4 lo, hi;
                lo.x = (r[0] & 0xffffu) | (r[1] << 16);
                lo.y = (r[2] & 0xffffu) | (r[3] << 16);
                lo.z = (r[4] & 0xffffu) | (r[5] << 16);
                lo.w = (r[6] & 0xffffu) | (r[7] << 16);
                hi.x = (r[0] >> 16) | (r[1] & 0xffff0000u);
                hi.y = (r[2] >> 16) | (r[3] & 0xffff0000u);
                hi.z = (r[4] >> 16) | (r[5] & 0xffff0000u);
                hi.w = (r[6] >> 16) | (r[7] & 0xffff0000u);
                *(uint4*)&Bst[tcol][trow]     = lo;
                *(uint4*)&Bst[tcol + 1][trow] = hi;
            } else {
                #pragma unroll
                for (int c = 0; c < 2; ++c) {
                    const int gn = n0 + tcol + c;
                    #pragma unroll
                    for (int e = 0; e < 8; ++e)
                        Bst[tcol + c][trow + e] =
                            (gn < N) ? src[(size_t)e * ldb + c] : (__bf16)0.0f;
                }
            }
        }
        __syncthreads();

        // ---- build fragments (all contiguous 16B LDS reads) ----
        // A (16x32): lane m = lane&15; elem e -> K = 8*halfsel + (e&7) + 16*(e>=8)
        v16bf afr[2];
        #pragma unroll
        for (int mi = 0; mi < 2; ++mi) {
            const int r = waveM * 32 + mi * 16 + sublane;
            FragU u;
            u.q[0] = *(const uint4*)&As[r][halfsel * 8];
            u.q[1] = *(const uint4*)&As[r][16 + halfsel * 8];
            afr[mi] = u.v;
        }
        // B (32x16): lane n = lane&15; elem e -> K = 8*halfsel + (e&7) + 16*(e>=8)
        v16bf bfr[2];
        #pragma unroll
        for (int ni = 0; ni < 2; ++ni) {
            const int cN = waveN * 32 + ni * 16 + sublane;
            FragU u;
            u.q[0] = *(const uint4*)&Bst[cN][halfsel * 8];
            u.q[1] = *(const uint4*)&Bst[cN][16 + halfsel * 8];
            bfr[ni] = u.v;
        }

        #pragma unroll
        for (int mi = 0; mi < 2; ++mi)
            #pragma unroll
            for (int ni = 0; ni < 2; ++ni)
                acc[mi][ni] = __builtin_amdgcn_wmma_f32_16x16x32_bf16(
                    false, afr[mi], false, bfr[ni],
                    (short)0, acc[mi][ni], false, false);

        __syncthreads();
    }

    // ---- epilogue ----
    // C/D (16x16 f32): VGPR g -> row (g + 8*(lane>=16)), col = lane&15
    float*          Cfp = Cf ? Cf + (size_t)batch * (size_t)sCf : nullptr;
    __hip_bfloat16* Cbp = Cb ? Cb + (size_t)batch * (size_t)sCb : nullptr;
    const float*    rp  = res ? res + (size_t)batch * (size_t)sR : nullptr;

    #pragma unroll
    for (int mi = 0; mi < 2; ++mi) {
        #pragma unroll
        for (int ni = 0; ni < 2; ++ni) {
            const int n = n0 + waveN * 32 + ni * 16 + sublane;
            if (n >= N) continue;
            const float bv = bias ? bias[n] : 0.0f;
            #pragma unroll
            for (int g = 0; g < 8; ++g) {
                const int m = m0 + waveM * 32 + mi * 16 + halfsel * 8 + g;
                if (m >= M) continue;
                float v = acc[mi][ni][g] + bv;
                if (rp) v += rp[(size_t)m * ldr + n];
                if (flags & 1) v = fmaxf(v, 0.0f);
                if (Cfp) Cfp[(size_t)m * ldcf + n] = v;
                if (Cbp) Cbp[(size_t)m * ldcb + n] = __float2bfloat16(v);
            }
        }
    }
}

// ---------------------------------------------------------------------------
// fp32 -> bf16 elementwise convert
// ---------------------------------------------------------------------------
__global__ void cvt_bf16_kernel(const float* __restrict__ src,
                                __hip_bfloat16* __restrict__ dst, int n)
{
    for (int i = blockIdx.x * 256 + threadIdx.x; i < n; i += gridDim.x * 256)
        dst[i] = __float2bfloat16(src[i]);
}

// ---------------------------------------------------------------------------
// LayerNorm (biased var, eps=1e-3) over last dim, one block per row -> bf16
// ---------------------------------------------------------------------------
__global__ __launch_bounds__(256)
void layernorm_kernel(const float* __restrict__ x, const float* __restrict__ g,
                      const float* __restrict__ b, __hip_bfloat16* __restrict__ out,
                      int C)
{
    __shared__ float red[256];
    const int row = blockIdx.x, tid = threadIdx.x;
    const float* xr = x + (size_t)row * C;

    float s = 0.0f;
    for (int c = tid; c < C; c += 256) s += xr[c];
    red[tid] = s; __syncthreads();
    for (int st = 128; st > 0; st >>= 1) {
        if (tid < st) red[tid] += red[tid + st];
        __syncthreads();
    }
    const float mean = red[0] / (float)C;
    __syncthreads();

    float sq = 0.0f;
    for (int c = tid; c < C; c += 256) { float d = xr[c] - mean; sq += d * d; }
    red[tid] = sq; __syncthreads();
    for (int st = 128; st > 0; st >>= 1) {
        if (tid < st) red[tid] += red[tid + st];
        __syncthreads();
    }
    const float var = red[0] / (float)C;
    const float rs  = rsqrtf(var + 1e-3f);

    __hip_bfloat16* orow = out + (size_t)row * C;
    for (int c = tid; c < C; c += 256)
        orow[c] = __float2bfloat16((xr[c] - mean) * rs * g[c] + b[c]);
}

// ---------------------------------------------------------------------------
// q [T,512] f32 -> qw/qr [H,T,64] bf16 with scale 1/sqrt(64) and rel biases
// ---------------------------------------------------------------------------
__global__ void qbias_kernel(const float* __restrict__ q,
                             const float* __restrict__ rwb,
                             const float* __restrict__ rrb,
                             __hip_bfloat16* __restrict__ qw,
                             __hip_bfloat16* __restrict__ qr, int T)
{
    const int idx = blockIdx.x * 256 + threadIdx.x;
    if (idx >= T * 512) return;
    const int c = idx & 511, t = idx >> 9;
    const int h = c >> 6, kk = c & 63;
    const float v = q[idx] * 0.125f;               // K^-0.5, K=64
    const size_t o = ((size_t)h * T + t) * 64 + kk;
    qw[o] = __float2bfloat16(v + rwb[c]);
    qr[o] = __float2bfloat16(v + rrb[c]);
}

// ---------------------------------------------------------------------------
// src [L,512] f32 -> dst [8,64,L] bf16 (per-head transpose for B operands)
// ---------------------------------------------------------------------------
__global__ void head_transpose_kernel(const float* __restrict__ src,
                                      __hip_bfloat16* __restrict__ dst, int L)
{
    const int idx = blockIdx.x * 256 + threadIdx.x;    // 512*L total
    if (idx >= 512 * L) return;
    const int m = idx % L;
    const int hk = idx / L;                            // h*64 + kk
    dst[idx] = __float2bfloat16(src[(size_t)m * 512 + hk]);
}

// ---------------------------------------------------------------------------
// Positional features: one wave32 per position p in [0, 2T-1).
// lane j = feature index within each of the 3 basis classes (n = F/6 = 32).
// ---------------------------------------------------------------------------
__global__ void posfeat_kernel(__hip_bfloat16* __restrict__ out, int T)
{
    const int p = blockIdx.x;
    const int j = threadIdx.x;                   // 0..31
    const float dist = (float)(p - (T - 1));
    const float ap   = fabsf(dist);
    const float fT   = (float)T;

    // exponential half-life decay: hl = 2^linspace(3, log2(T), 32)
    const float hl = exp2f(3.0f + (float)j * (log2f(fT) - 3.0f) / 31.0f);
    const float fe = exp2f(-ap / hl);
    // central mask
    const float width = exp2f((float)(j + 1)) - 1.0f;
    const float fc = (width > ap) ? 1.0f : 0.0f;
    // gamma pdf bumps
    const float stddev = fT / 64.0f;                       // T/(2n)
    const float mean   = fT / 32.0f + (float)j * (fT - fT / 32.0f) / 31.0f;
    const float conc   = (mean / stddev) * (mean / stddev);
    const float rate   = mean / (stddev * stddev);
    const float lu = (conc - 1.0f) * logf(ap) - rate * ap; // logf(0)=-inf -> exp=0
    const float ln = lgammaf(conc) - conc * logf(rate);
    const float prob = expf(lu - ln) + 1e-8f;
    float mx = prob;
    for (int o = 16; o > 0; o >>= 1) mx = fmaxf(mx, __shfl_xor(mx, o, 32));
    const float fg = prob / mx;

    const float sg = (dist > 0.0f) ? 1.0f : ((dist < 0.0f) ? -1.0f : 0.0f);
    const size_t base = (size_t)p * 192;
    out[base + j]        = __float2bfloat16(fe);
    out[base + 32 + j]   = __float2bfloat16(fc);
    out[base + 64 + j]   = __float2bfloat16(fg);
    out[base + 96 + j]   = __float2bfloat16(sg * fe);
    out[base + 128 + j]  = __float2bfloat16(sg * fc);
    out[base + 160 + j]  = __float2bfloat16(sg * fg);
}

// ---------------------------------------------------------------------------
// Fused relative-shift + softmax:
//   logits[h,q,m] = content[h,q,m] + rel[h,q, m - q + T - 1]
// One block per (q,h) row; T = 1536 = 6*256. Writes attn in bf16.
// ---------------------------------------------------------------------------
__global__ __launch_bounds__(256)
void softmax_kernel(const float* __restrict__ content,
                    const float* __restrict__ rel,
                    __hip_bfloat16* __restrict__ attn, int T, int W)
{
    __shared__ float red[256];
    const int q = blockIdx.x, h = blockIdx.y, tid = threadIdx.x;
    const float* cr = content + ((size_t)h * T + q) * T;
    const float* rr = rel + ((size_t)h * T + q) * W + (T - 1 - q);

    float l[6];
    float mx = -3.4e38f;
    #pragma unroll
    for (int i = 0; i < 6; ++i) {
        const int m = tid + i * 256;
        l[i] = cr[m] + rr[m];
        mx = fmaxf(mx, l[i]);
    }
    red[tid] = mx; __syncthreads();
    for (int st = 128; st > 0; st >>= 1) {
        if (tid < st) red[tid] = fmaxf(red[tid], red[tid + st]);
        __syncthreads();
    }
    mx = red[0]; __syncthreads();

    float s = 0.0f;
    #pragma unroll
    for (int i = 0; i < 6; ++i) { l[i] = expf(l[i] - mx); s += l[i]; }
    red[tid] = s; __syncthreads();
    for (int st = 128; st > 0; st >>= 1) {
        if (tid < st) red[tid] += red[tid + st];
        __syncthreads();
    }
    const float inv = 1.0f / red[0];

    __hip_bfloat16* ar = attn + ((size_t)h * T + q) * T;
    #pragma unroll
    for (int i = 0; i < 6; ++i)
        ar[tid + i * 256] = __float2bfloat16(l[i] * inv);
}

// ---------------------------------------------------------------------------
// Host side
// ---------------------------------------------------------------------------
typedef __hip_bfloat16 bf16_t;

static void launch_gemm(hipStream_t s,
                        const bf16_t* A, long long sA, int lda,
                        const bf16_t* B, long long sB, int ldb,
                        float* Cf, long long sCf, int ldcf,
                        bf16_t* Cb, long long sCb, int ldcb,
                        const float* bias,
                        const float* res, long long sR, int ldr,
                        int M, int N, int Kd, int flags, int batch)
{
    dim3 grid((N + BN - 1) / BN, (M + BM - 1) / BM, batch);
    gemm_bf16_kernel<<<grid, 256, 0, s>>>(A, sA, lda, B, sB, ldb,
                                          Cf, sCf, ldcf, Cb, sCb, ldcb,
                                          bias, res, sR, ldr, M, N, Kd, flags);
}

extern "C" void kernel_launch(void* const* d_in, const int* in_sizes, int n_in,
                              void* d_out, int out_size, void* d_ws, size_t ws_size,
                              hipStream_t stream)
{
    (void)in_sizes; (void)n_in; (void)out_size; (void)ws_size;
    const int T = 1536, C = 1536, H = 8;
    const int W = 2 * T - 1;     // 3071

    const float* x    = (const float*)d_in[0];
    const float* ln1g = (const float*)d_in[1];
    const float* ln1b = (const float*)d_in[2];
    const float* ln2g = (const float*)d_in[3];
    const float* ln2b = (const float*)d_in[4];
    const float* Wq   = (const float*)d_in[5];
    const float* Wk   = (const float*)d_in[6];
    const float* Wv   = (const float*)d_in[7];
    const float* Wr   = (const float*)d_in[8];
    const float* Wo   = (const float*)d_in[9];
    const float* bo   = (const float*)d_in[10];
    const float* rwb  = (const float*)d_in[11];
    const float* rrb  = (const float*)d_in[12];
    const float* W1   = (const float*)d_in[13];
    const float* b1   = (const float*)d_in[14];
    const float* W2   = (const float*)d_in[15];
    const float* b2   = (const float*)d_in[16];

    // ---- workspace carve-out ----
    char* base = (char*)d_ws;
    size_t off = 0;
    auto alloc = [&](size_t bytes) -> void* {
        off = (off + 255) & ~(size_t)255;
        void* p = base + off;
        off += bytes;
        return p;
    };
    bf16_t* wq_b  = (bf16_t*)alloc((size_t)C * 512 * 2);
    bf16_t* wk_b  = (bf16_t*)alloc((size_t)C * 512 * 2);
    bf16_t* wv_b  = (bf16_t*)alloc((size_t)C * C * 2);
    bf16_t* wr_b  = (bf16_t*)alloc((size_t)192 * 512 * 2);
    bf16_t* wo_b  = (bf16_t*)alloc((size_t)C * C * 2);
    bf16_t* w1_b  = (bf16_t*)alloc((size_t)C * 3072 * 2);
    bf16_t* w2_b  = (bf16_t*)alloc((size_t)3072 * C * 2);
    bf16_t* xn_b  = (bf16_t*)alloc((size_t)T * C * 2);
    float*  q_f   = (float*) alloc((size_t)T * 512 * 4);
    float*  k_f   = (float*) alloc((size_t)T * 512 * 4);
    bf16_t* qw_b  = (bf16_t*)alloc((size_t)H * T * 64 * 2);
    bf16_t* qr_b  = (bf16_t*)alloc((size_t)H * T * 64 * 2);
    bf16_t* kT_b  = (bf16_t*)alloc((size_t)H * 64 * T * 2);
    bf16_t* v_b   = (bf16_t*)alloc((size_t)T * C * 2);
    bf16_t* pos_b = (bf16_t*)alloc((size_t)W * 192 * 2);
    float*  rk_f  = (float*) alloc((size_t)W * 512 * 4);
    bf16_t* rT_b  = (bf16_t*)alloc((size_t)H * 64 * W * 2);
    float*  cont  = (float*) alloc((size_t)H * T * T * 4);
    float*  rel   = (float*) alloc((size_t)H * T * W * 4);
    bf16_t* at_b  = (bf16_t*)alloc((size_t)H * T * T * 2);
    bf16_t* o_b   = (bf16_t*)alloc((size_t)T * C * 2);
    float*  y_f   = (float*) alloc((size_t)T * C * 4);
    bf16_t* yn_b  = (bf16_t*)alloc((size_t)T * C * 2);
    bf16_t* h1_b  = (bf16_t*)alloc((size_t)T * 3072 * 2);

    auto cvt = [&](const float* s, bf16_t* d, int n) {
        int blocks = (n + 255) / 256; if (blocks > 4096) blocks = 4096;
        cvt_bf16_kernel<<<blocks, 256, 0, stream>>>(s, d, n);
    };
    // 1) weights -> bf16
    cvt(Wq, wq_b, C * 512);
    cvt(Wk, wk_b, C * 512);
    cvt(Wv, wv_b, C * C);
    cvt(Wr, wr_b, 192 * 512);
    cvt(Wo, wo_b, C * C);
    cvt(W1, w1_b, C * 3072);
    cvt(W2, w2_b, 3072 * C);

    // 2) LN1
    layernorm_kernel<<<T, 256, 0, stream>>>(x, ln1g, ln1b, xn_b, C);

    // 3) Q, K (f32), V (bf16 direct)
    launch_gemm(stream, xn_b, 0, C, wq_b, 0, 512, q_f, 0, 512,
                nullptr, 0, 0, nullptr, nullptr, 0, 0, T, 512, C, 0, 1);
    launch_gemm(stream, xn_b, 0, C, wk_b, 0, 512, k_f, 0, 512,
                nullptr, 0, 0, nullptr, nullptr, 0, 0, T, 512, C, 0, 1);
    launch_gemm(stream, xn_b, 0, C, wv_b, 0, C, nullptr, 0, 0,
                v_b, 0, C, nullptr, nullptr, 0, 0, T, C, C, 0, 1);

    // 4) q*scale + biases -> qw/qr [H,T,64]; kT [H,64,T]
    qbias_kernel<<<(T * 512 + 255) / 256, 256, 0, stream>>>(q_f, rwb, rrb, qw_b, qr_b, T);
    head_transpose_kernel<<<(512 * T + 255) / 256, 256, 0, stream>>>(k_f, kT_b, T);

    // 5) positional features -> pos [W,192] bf16; r_k = pos @ Wr; rT [H,64,W]
    posfeat_kernel<<<W, 32, 0, stream>>>(pos_b, T);
    launch_gemm(stream, pos_b, 0, 192, wr_b, 0, 512, rk_f, 0, 512,
                nullptr, 0, 0, nullptr, nullptr, 0, 0, W, 512, 192, 0, 1);
    head_transpose_kernel<<<(512 * W + 255) / 256, 256, 0, stream>>>(rk_f, rT_b, W);

    // 6) content[h] = qw[h] @ kT[h]   (batched over heads)
    launch_gemm(stream, qw_b, (long long)T * 64, 64,
                kT_b, (long long)64 * T, T,
                cont, (long long)T * T, T,
                nullptr, 0, 0, nullptr, nullptr, 0, 0, T, T, 64, 0, H);

    // 7) rel[h] = qr[h] @ rT[h]  [T x W]
    launch_gemm(stream, qr_b, (long long)T * 64, 64,
                rT_b, (long long)64 * W, W,
                rel, (long long)T * W, W,
                nullptr, 0, 0, nullptr, nullptr, 0, 0, T, W, 64, 0, H);

    // 8) fused relative-shift + softmax -> attn bf16
    {
        dim3 g(T, H, 1);
        softmax_kernel<<<g, 256, 0, stream>>>(cont, rel, at_b, T, W);
    }

    // 9) o[h] = attn[h] @ v[:, h*192:(h+1)*192]  -> o_b [T, H*192] bf16
    launch_gemm(stream, at_b, (long long)T * T, T,
                v_b + 0, 192, C,                    // B batch stride 192 cols
                nullptr, 0, 0,
                o_b + 0, 192, C,                    // C batch stride 192 cols
                nullptr, nullptr, 0, 0, T, 192, T, 0, H);

    // 10) y = x + o @ Wo + bo
    launch_gemm(stream, o_b, 0, C, wo_b, 0, C, y_f, 0, C,
                nullptr, 0, 0, bo, x, 0, C, T, C, C, 0, 1);

    // 11) LN2
    layernorm_kernel<<<T, 256, 0, stream>>>(y_f, ln2g, ln2b, yn_b, C);

    // 12) h1 = relu(yn @ W1 + b1)  (bf16)
    launch_gemm(stream, yn_b, 0, C, w1_b, 0, 3072, nullptr, 0, 0,
                h1_b, 0, 3072, b1, nullptr, 0, 0, T, 3072, C, 1, 1);

    // 13) out = y + h1 @ W2 + b2
    launch_gemm(stream, h1_b, 0, 3072, w2_b, 0, C, (float*)d_out, 0, C,
                nullptr, 0, 0, b2, y_f, 0, C, T, C, 3072, 0, 1);
}
